// ElmanMamba2_86466281603870
// MI455X (gfx1250) — matrix operand order repaired and checked by reading, not compiled, not run on hardware
//
#include <hip/hip_runtime.h>
#include <hip/hip_bf16.h>

// ------------------------------------------------------------------
// ElmanMamba2 on MI455X (gfx1250):
//   kernel 0 (x3): one-time f32 -> bf16(hi)+bf16(lo) split of x, Wd, Wb
//   kernel 1: fused dual-GEMM (x@Wd^T, x@Wb^T) via bf16x3 split WMMA,
//             register double-buffered over K (SW pipeline), epilogue
//             a = sigmoid(-(zd+bd)) [= exp(-softplus)], bb=(1-a)*(zb+bB)
//   kernel 2: per-chunk scan summaries (P = prod a, S = local scan)
//   kernel 3: serial scan over 64 chunk summaries -> chunk carries
//   kernel 4: replay each chunk from its carry, write hs into d_out
// ------------------------------------------------------------------

#define T_DIM 2048
#define B_DIM 8
#define D_DIM 1024
#define H_DIM 1024
#define M_DIM (T_DIM * B_DIM)      // 16384
#define BH    (B_DIM * H_DIM)      // 8192
#define NCHUNK 64
#define CHUNK_T 32                 // 64*32 = 2048

typedef __attribute__((ext_vector_type(16))) __bf16 v16bf;
typedef __attribute__((ext_vector_type(8)))  float  v8f;

union Frag {
  v16bf v;
  uint4 q[2];
};

__device__ __forceinline__ unsigned pack_hi_pair(float a, float b) {
  // truncate-to-bf16 of both floats, packed (a in [15:0], b in [31:16])
  return (__float_as_uint(b) & 0xFFFF0000u) | (__float_as_uint(a) >> 16);
}
__device__ __forceinline__ float bf16_resid(float a) {
  return a - __uint_as_float(__float_as_uint(a) & 0xFFFF0000u);
}

// --------------------------------------------------------------
// Kernel 0: split f32 array into bf16 hi / bf16 lo arrays.
// --------------------------------------------------------------
__global__ void split_bf16_kernel(const float* __restrict__ src,
                                  uint4* __restrict__ hi,
                                  uint4* __restrict__ lo,
                                  int n8) {
  const int g = blockIdx.x * 256 + threadIdx.x;
  if (g >= n8) return;
  const float4* s = (const float4*)src + 2 * (size_t)g;
  const float4 f0 = s[0];
  const float4 f1 = s[1];
  const float fs[8] = {f0.x, f0.y, f0.z, f0.w, f1.x, f1.y, f1.z, f1.w};
  unsigned h[4], l[4];
#pragma unroll
  for (int j = 0; j < 4; ++j) {
    const float a = fs[2 * j], b = fs[2 * j + 1];
    h[j] = pack_hi_pair(a, b);
    l[j] = pack_hi_pair(bf16_resid(a), bf16_resid(b));
  }
  hi[g] = make_uint4(h[0], h[1], h[2], h[3]);
  lo[g] = make_uint4(l[0], l[1], l[2], l[3]);
}

// A fragment (16x32 bf16, MxK): lane half kh holds K = {8kh..8kh+7} U {16+8kh..}
__device__ __forceinline__ void loadFragA(const unsigned short* __restrict__ p, Frag& f) {
  f.q[0] = *(const uint4*)(p);
  f.q[1] = *(const uint4*)(p + 16);
}

// B fragment (32x16 bf16, KxN): lane half kh holds K = {8kh..8kh+7} U {16+8kh..}
__device__ __forceinline__ void loadFragB(const unsigned short* __restrict__ p, Frag& f) {
  f.q[0] = *(const uint4*)(p);
  f.q[1] = *(const uint4*)(p + 16);
}

// 3-term split-precision bf16 MMA: ~fp32 accuracy (drops lo*lo term)
__device__ __forceinline__ v8f mma3(v8f acc, const Frag& ah, const Frag& al,
                                    const Frag& bh, const Frag& bl) {
  acc = __builtin_amdgcn_wmma_f32_16x16x32_bf16(false, ah.v, false, bh.v, (short)0, acc, false, false);
  acc = __builtin_amdgcn_wmma_f32_16x16x32_bf16(false, ah.v, false, bl.v, (short)0, acc, false, false);
  acc = __builtin_amdgcn_wmma_f32_16x16x32_bf16(false, al.v, false, bh.v, (short)0, acc, false, false);
  return acc;
}

// One K-chunk (K=32) worth of operand fragments: 16 Frags = 128 VGPRs
struct KBuf {
  Frag ah[4], al[4];          // A hi/lo for 4 M-subtiles
  Frag dh[2], dl[2];          // W_delta hi/lo for 2 N-subtiles
  Frag gh[2], gl[2];          // W_B     hi/lo for 2 N-subtiles
};

__device__ __forceinline__ void loadBuf(
    KBuf& b, int kb,
    const unsigned short* __restrict__ xh, const unsigned short* __restrict__ xl,
    const unsigned short* __restrict__ wdh, const unsigned short* __restrict__ wdl,
    const unsigned short* __restrict__ wbh, const unsigned short* __restrict__ wbl,
    const size_t (&aOff)[4], const size_t (&bOff)[2]) {
#pragma unroll
  for (int i = 0; i < 4; ++i) {
    loadFragA(xh + aOff[i] + kb, b.ah[i]);
    loadFragA(xl + aOff[i] + kb, b.al[i]);
  }
#pragma unroll
  for (int j = 0; j < 2; ++j) {
    loadFragB(wdh + bOff[j] + kb, b.dh[j]);
    loadFragB(wdl + bOff[j] + kb, b.dl[j]);
    loadFragB(wbh + bOff[j] + kb, b.gh[j]);
    loadFragB(wbl + bOff[j] + kb, b.gl[j]);
  }
}

__device__ __forceinline__ void mmaBuf(const KBuf& b,
                                       v8f (&accD)[4][2], v8f (&accB)[4][2]) {
#pragma unroll
  for (int i = 0; i < 4; ++i)
#pragma unroll
    for (int j = 0; j < 2; ++j) {
      accD[i][j] = mma3(accD[i][j], b.ah[i], b.al[i], b.dh[j], b.dl[j]);
      accB[i][j] = mma3(accB[i][j], b.ah[i], b.al[i], b.gh[j], b.gl[j]);
    }
}

// --------------------------------------------------------------
// Kernel 1: fused dual projection GEMM + gating epilogue
// grid: (H/128, M/128), block: 256 threads = 8 wave32
// wave grid inside block: 2 (M) x 4 (N); each wave: 64(M) x 32(N)
// K loop double-buffered in registers (two KBufs in flight)
// --------------------------------------------------------------
__global__ __launch_bounds__(256, 1) void proj_gemm_kernel(
    const unsigned short* __restrict__ xh, const unsigned short* __restrict__ xl,
    const unsigned short* __restrict__ wdh, const unsigned short* __restrict__ wdl,
    const unsigned short* __restrict__ wbh, const unsigned short* __restrict__ wbl,
    const float* __restrict__ bd, const float* __restrict__ bB,
    float* __restrict__ wsA, float* __restrict__ outBb)
{
  const int lane = threadIdx.x & 31;
  const int wave = threadIdx.x >> 5;
  const int wm = wave & 1;        // 0..1 -> M offset 0,64
  const int wn = wave >> 1;       // 0..3 -> N offset 0,32,64,96
  const int kh  = lane >> 4;      // lane half (K range select)
  const int l15 = lane & 15;
  const int mBase = blockIdx.y * 128 + wm * 64;
  const int nBase = blockIdx.x * 128 + wn * 32;

  size_t aOff[4];
#pragma unroll
  for (int i = 0; i < 4; ++i)
    aOff[i] = (size_t)(mBase + i * 16 + l15) * D_DIM + 8 * kh;
  size_t bOff[2];
#pragma unroll
  for (int j = 0; j < 2; ++j)
    bOff[j] = (size_t)(nBase + j * 16 + l15) * D_DIM + 8 * kh;

  v8f accD[4][2], accB[4][2];
  const v8f vzero = {0.f, 0.f, 0.f, 0.f, 0.f, 0.f, 0.f, 0.f};
#pragma unroll
  for (int i = 0; i < 4; ++i)
#pragma unroll
    for (int j = 0; j < 2; ++j) { accD[i][j] = vzero; accB[i][j] = vzero; }

  KBuf buf0, buf1;
  loadBuf(buf0, 0, xh, xl, wdh, wdl, wbh, wbl, aOff, bOff);

#pragma unroll 1
  for (int kb = 0; kb < D_DIM - 64; kb += 64) {
    loadBuf(buf1, kb + 32, xh, xl, wdh, wdl, wbh, wbl, aOff, bOff);
    // warm WGP$ two chunks ahead on the A stream
#pragma unroll
    for (int i = 0; i < 4; ++i)
      __builtin_prefetch(xh + aOff[i] + kb + 64, 0, 3);
    mmaBuf(buf0, accD, accB);
    loadBuf(buf0, kb + 64, xh, xl, wdh, wdl, wbh, wbl, aOff, bOff);
    mmaBuf(buf1, accD, accB);
  }
  // tail: kb == D_DIM-64; buf0 holds chunk D_DIM-64
  loadBuf(buf1, D_DIM - 32, xh, xl, wdh, wdl, wbh, wbl, aOff, bOff);
  mmaBuf(buf0, accD, accB);
  mmaBuf(buf1, accD, accB);

  // Epilogue: a = 1/(1+exp(v)) == exp(-softplus(v)); bb = (1-a)*(zB+bB)
#pragma unroll
  for (int j = 0; j < 2; ++j) {
    const int n = nBase + j * 16 + l15;
    const float bdv = bd[n];
    const float bBv = bB[n];
#pragma unroll
    for (int i = 0; i < 4; ++i) {
#pragma unroll
      for (int r = 0; r < 8; ++r) {
        const int m = mBase + i * 16 + r + 8 * kh;   // C/D layout: VGPR r -> M=r(+8)
        const float vd = accD[i][j][r] + bdv;
        const float vb = accB[i][j][r] + bBv;
        const float a  = 1.0f / (1.0f + __expf(vd));
        const size_t idx = (size_t)m * H_DIM + n;
        wsA[idx]   = a;
        outBb[idx] = (1.0f - a) * vb;
      }
    }
  }
}

// --------------------------------------------------------------
// Kernel 2: per-chunk summaries.  h_out = P*h_in + S over 32 steps
// --------------------------------------------------------------
__global__ void chunk_summary_kernel(const float* __restrict__ wsA,
                                     const float* __restrict__ outBb,
                                     float* __restrict__ P, float* __restrict__ S)
{
  const int i = (blockIdx.x & 31) * 256 + threadIdx.x;  // 0..8191 (b,h)
  const int c = blockIdx.x >> 5;                        // chunk
  float p = 1.0f, s = 0.0f;
  size_t idx = (size_t)c * CHUNK_T * BH + i;
#pragma unroll 4
  for (int j = 0; j < CHUNK_T; ++j, idx += BH) {
    const float a = wsA[idx];
    const float b = outBb[idx];
    s = fmaf(a, s, b);
    p *= a;
  }
  P[(size_t)c * BH + i] = p;
  S[(size_t)c * BH + i] = s;
}

// --------------------------------------------------------------
// Kernel 3: serial scan over chunk summaries -> per-chunk carry-in
// --------------------------------------------------------------
__global__ void chunk_scan_kernel(const float* __restrict__ P,
                                  const float* __restrict__ S,
                                  const float* __restrict__ h0,
                                  float* __restrict__ Cin)
{
  const int i = blockIdx.x * 256 + threadIdx.x;
  float carry = h0[i];
#pragma unroll 4
  for (int c = 0; c < NCHUNK; ++c) {
    const size_t idx = (size_t)c * BH + i;
    Cin[idx] = carry;
    carry = fmaf(P[idx], carry, S[idx]);
  }
}

// --------------------------------------------------------------
// Kernel 4: replay chunk from correct carry; write hs (in place on
// d_out, which currently holds bb; same-thread read-then-write)
// --------------------------------------------------------------
__global__ void fixup_kernel(const float* __restrict__ wsA,
                             const float* __restrict__ Cin,
                             float* __restrict__ out)
{
  const int i = (blockIdx.x & 31) * 256 + threadIdx.x;
  const int c = blockIdx.x >> 5;
  float h = Cin[(size_t)c * BH + i];
  size_t idx = (size_t)c * CHUNK_T * BH + i;
#pragma unroll 4
  for (int j = 0; j < CHUNK_T; ++j, idx += BH) {
    h = fmaf(wsA[idx], h, out[idx]);
    out[idx] = h;
  }
}

extern "C" void kernel_launch(void* const* d_in, const int* in_sizes, int n_in,
                              void* d_out, int out_size, void* d_ws, size_t ws_size,
                              hipStream_t stream) {
  const float* x  = (const float*)d_in[0];   // [T,B,D]
  const float* h0 = (const float*)d_in[1];   // [B,H]
  const float* Wd = (const float*)d_in[2];   // [H,D]
  const float* bd = (const float*)d_in[3];   // [H]
  const float* Wb = (const float*)d_in[4];   // [H,D]
  const float* bB = (const float*)d_in[5];   // [H]
  float* out = (float*)d_out;                // [T,B,H]

  // workspace layout (all 16B aligned):
  char* ws = (char*)d_ws;
  float* wsA = (float*)ws;                                  // 64 MB  decay a
  float* P   = wsA + (size_t)M_DIM * H_DIM;                 // 2 MB
  float* S   = P   + (size_t)NCHUNK * BH;                   // 2 MB
  float* Cin = S   + (size_t)NCHUNK * BH;                   // 2 MB
  unsigned short* xh  = (unsigned short*)(Cin + (size_t)NCHUNK * BH);  // 32 MB
  unsigned short* xl  = xh  + (size_t)M_DIM * D_DIM;        // 32 MB
  unsigned short* wdh = xl  + (size_t)M_DIM * D_DIM;        // 2 MB
  unsigned short* wdl = wdh + (size_t)H_DIM * D_DIM;        // 2 MB
  unsigned short* wbh = wdl + (size_t)H_DIM * D_DIM;        // 2 MB
  unsigned short* wbl = wbh + (size_t)H_DIM * D_DIM;        // 2 MB

  // ---- pass 0: one-time f32 -> bf16 hi/lo splits ----
  const int n8x = M_DIM * D_DIM / 8;   // 2M groups
  const int n8w = H_DIM * D_DIM / 8;   // 128K groups
  split_bf16_kernel<<<n8x / 256, 256, 0, stream>>>(x,  (uint4*)xh,  (uint4*)xl,  n8x);
  split_bf16_kernel<<<n8w / 256, 256, 0, stream>>>(Wd, (uint4*)wdh, (uint4*)wdl, n8w);
  split_bf16_kernel<<<n8w / 256, 256, 0, stream>>>(Wb, (uint4*)wbh, (uint4*)wbl, n8w);

  // ---- pass 1: WMMA dual-GEMM + gating epilogue ----
  dim3 g1(H_DIM / 128, M_DIM / 128);   // (8, 128)
  proj_gemm_kernel<<<g1, 256, 0, stream>>>(xh, xl, wdh, wdl, wbh, wbl,
                                           bd, bB, wsA, out);

  // ---- passes 2-4: chunked parallel linear recurrence ----
  chunk_summary_kernel<<<NCHUNK * (BH / 256), 256, 0, stream>>>(wsA, out, P, S);
  chunk_scan_kernel<<<BH / 256, 256, 0, stream>>>(P, S, h0, Cin);
  fixup_kernel<<<NCHUNK * (BH / 256), 256, 0, stream>>>(wsA, Cin, out);
}
